// SelectiveStateSpaceModel_884763263715
// MI455X (gfx1250) — matrix operand order repaired and checked by reading, not compiled, not run on hardware
//
#include <hip/hip_runtime.h>
#include <hip/hip_bf16.h>

#define D_INNER 2048
#define DT_RANK 64
#define D_STATE 16
#define SEQ_LEN 2048
#define N_DBC   (DT_RANK + 2 * D_STATE)   // 96

typedef __attribute__((ext_vector_type(16))) _Float16 v16h;
typedef __attribute__((ext_vector_type(8)))  _Float16 v8h;
typedef __attribute__((ext_vector_type(8)))  float    v8f;

// ---------------------------------------------------------------------------
// Kernel 0: one-shot f32 -> f16 conversion of x, W_in, W_dt (WMMA operands)
// ---------------------------------------------------------------------------
__global__ __launch_bounds__(256)
void ssm_cvt_f16_kernel(const float* __restrict__ x,
                        const float* __restrict__ w_in,
                        const float* __restrict__ w_dt,
                        _Float16* __restrict__ xh,
                        _Float16* __restrict__ winh,
                        _Float16* __restrict__ wdth) {
  const int nx = SEQ_LEN * D_INNER;       // 4,194,304
  const int nw = N_DBC * D_INNER;         //   196,608
  const int nd = D_INNER * DT_RANK;       //   131,072
  const int stride = gridDim.x * blockDim.x;
  for (int i = blockIdx.x * blockDim.x + threadIdx.x; i < nx; i += stride)
    xh[i] = (_Float16)x[i];
  for (int i = blockIdx.x * blockDim.x + threadIdx.x; i < nw; i += stride)
    winh[i] = (_Float16)w_in[i];
  for (int i = blockIdx.x * blockDim.x + threadIdx.x; i < nd; i += stride)
    wdth[i] = (_Float16)w_dt[i];
}

// ---------------------------------------------------------------------------
// Kernel 1: dbc[L,96] = x[L,2048] @ W_in[96,2048]^T  via v_wmma_f32_16x16x32_f16
// One 16x16 output tile per wave. 128 M-tiles x 6 N-tiles = 768 tiles.
// ---------------------------------------------------------------------------
__global__ __launch_bounds__(256)
void ssm_gemm_inproj_kernel(const _Float16* __restrict__ xh,    // [L, D_INNER]
                            const _Float16* __restrict__ winh,  // [96, D_INNER]
                            float*    __restrict__ dbc_f,       // [L, 96]
                            _Float16* __restrict__ dbc_h) {     // [L, 96]
  const int lane = threadIdx.x & 31;
  const int wave = threadIdx.x >> 5;
  const int tile = blockIdx.x * 8 + wave;      // 96 blocks * 8 waves = 768
  const int tm = tile / 6;
  const int tn = tile % 6;
  const int m  = lane & 15;
  const int hs = lane >> 4;

  // A: row (tm*16+m), half-selected k-base hs*16 (contiguous, matches B)
  const _Float16* pa = xh   + (size_t)(tm * 16 + m) * D_INNER + hs * 16;
  // B: column (tn*16 + (lane&15)), K group hs*16 (contiguous k, same order as A)
  const _Float16* pb = winh + (size_t)(tn * 16 + m) * D_INNER + hs * 16;

  v8f acc = {};
  for (int k0 = 0; k0 < D_INNER; k0 += 32) {
    v8h alo = *(const v8h*)(pa + k0);        // K = kbase + 0..7
    v8h ahi = *(const v8h*)(pa + k0 + 8);    // K = kbase + 8..15
    v16h a;
#pragma unroll
    for (int i = 0; i < 8; ++i) { a[i] = alo[i]; a[i + 8] = ahi[i]; }
    v16h b = *(const v16h*)(pb + k0);        // K = hs*16 + 0..15, col fixed
    acc = __builtin_amdgcn_wmma_f32_16x16x32_f16(false, a, false, b,
                                                 (short)0, acc, false, false);
  }
#pragma unroll
  for (int r = 0; r < 8; ++r) {              // C/D: VGPR r -> row r + hs*8
    const int row = tm * 16 + r + hs * 8;
    const int col = tn * 16 + m;
    const float v = acc[r];
    dbc_f[row * N_DBC + col] = v;
    dbc_h[row * N_DBC + col] = (_Float16)v;
  }
}

// ---------------------------------------------------------------------------
// Kernel 2: delta[L,2048] = softplus(dbc[:, :64] @ W_dt[2048,64]^T + b_dt)
// 128 x 128 tiles, K=64 (two WMMA steps per tile).
// ---------------------------------------------------------------------------
__global__ __launch_bounds__(256)
void ssm_gemm_delta_kernel(const _Float16* __restrict__ dbch,  // [L, 96]
                           const _Float16* __restrict__ wdth,  // [2048, 64]
                           const float*    __restrict__ b_dt,  // [2048]
                           float*          __restrict__ delta) // [L, 2048]
{
  const int lane = threadIdx.x & 31;
  const int wave = threadIdx.x >> 5;
  const int tile = blockIdx.x * 8 + wave;    // 2048 blocks * 8 = 16384 tiles
  const int tm = tile >> 7;                  // 0..127 (L tiles)
  const int tn = tile & 127;                 // 0..127 (d_inner tiles)
  const int m  = lane & 15;
  const int hs = lane >> 4;

  const _Float16* pa = dbch + (size_t)(tm * 16 + m) * N_DBC   + hs * 16;
  const _Float16* pb = wdth + (size_t)(tn * 16 + m) * DT_RANK + hs * 16;

  v8f acc = {};
#pragma unroll
  for (int k0 = 0; k0 < DT_RANK; k0 += 32) {
    v8h alo = *(const v8h*)(pa + k0);
    v8h ahi = *(const v8h*)(pa + k0 + 8);
    v16h a;
#pragma unroll
    for (int i = 0; i < 8; ++i) { a[i] = alo[i]; a[i + 8] = ahi[i]; }
    v16h b = *(const v16h*)(pb + k0);
    acc = __builtin_amdgcn_wmma_f32_16x16x32_f16(false, a, false, b,
                                                 (short)0, acc, false, false);
  }
#pragma unroll
  for (int r = 0; r < 8; ++r) {
    const int row = tm * 16 + r + hs * 8;    // l index
    const int d   = tn * 16 + m;             // channel index
    float v = acc[r] + b_dt[d];
    float sp = (v > 20.0f) ? v : log1pf(__expf(v));   // softplus
    delta[(size_t)row * D_INNER + d] = sp;
  }
}

// ---------------------------------------------------------------------------
// Kernel 3: selective scan. One state element per lane; each wave32 carries
// two channels (lanes 0-15 = channel c0 states s=0..15, lanes 16-31 = c1).
// 2048 channels -> 1024 waves -> 128 blocks of 8 waves.
// Per step: dA=exp(dt*A), h=dA*h+dt*B*x, y=reduce16(h*C)+x*D.
// ---------------------------------------------------------------------------
__global__ __launch_bounds__(256)
void ssm_scan_kernel(const float* __restrict__ x,       // [L, 2048]
                     const float* __restrict__ dbc_f,   // [L, 96] (B at 64, C at 80)
                     const float* __restrict__ delta,   // [L, 2048]
                     const float* __restrict__ A_log,   // [2048, 16]
                     const float* __restrict__ Dp,      // [2048]
                     float*       __restrict__ out)     // [L, 2048]
{
  const int lane = threadIdx.x & 31;
  const int wave = threadIdx.x >> 5;
  const int s  = lane & 15;
  const int hs = lane >> 4;
  const int ch = (blockIdx.x * 8 + wave) * 2 + hs;      // 0..2047

  const float As  = -__expf(A_log[ch * D_STATE + s]);   // A = -exp(A_log)
  const float Dch = Dp[ch];

  const float* pB = dbc_f + DT_RANK + s;                // B[l, s]
  const float* pC = dbc_f + DT_RANK + D_STATE + s;      // C[l, s]

  float h = 0.0f;
  for (int l = 0; l < SEQ_LEN; ++l) {
    const float dt = delta[(size_t)l * D_INNER + ch];   // broadcast in half-wave
    const float xv = x[(size_t)l * D_INNER + ch];
    const float Bv = pB[(size_t)l * N_DBC];
    const float Cv = pC[(size_t)l * N_DBC];

    if (l + 8 < SEQ_LEN) {                              // uniform branch
      __builtin_prefetch(&delta[(size_t)(l + 8) * D_INNER + ch], 0, 0);
      __builtin_prefetch(&x[(size_t)(l + 8) * D_INNER + ch], 0, 0);
    }

    const float dA = __expf(dt * As);
    h = __builtin_fmaf(dA, h, dt * Bv * xv);

    float c = h * Cv;                                   // reduce over 16 states
    c += __shfl_xor(c, 8, 32);
    c += __shfl_xor(c, 4, 32);
    c += __shfl_xor(c, 2, 32);
    c += __shfl_xor(c, 1, 32);
    if (s == 0)
      out[(size_t)l * D_INNER + ch] = __builtin_fmaf(xv, Dch, c);
  }
}

// ---------------------------------------------------------------------------
extern "C" void kernel_launch(void* const* d_in, const int* in_sizes, int n_in,
                              void* d_out, int out_size, void* d_ws, size_t ws_size,
                              hipStream_t stream) {
  const float* x     = (const float*)d_in[0];   // [2048, 2048]
  const float* W_in  = (const float*)d_in[1];   // [96, 2048]
  const float* W_dt  = (const float*)d_in[2];   // [2048, 64]
  const float* b_dt  = (const float*)d_in[3];   // [2048]
  const float* A_log = (const float*)d_in[4];   // [2048, 16]
  const float* D     = (const float*)d_in[5];   // [2048]
  float* out = (float*)d_out;

  char* ws = (char*)d_ws;
  size_t off = 0;
  auto carve = [&](size_t bytes) -> void* {
    void* p = ws + off;
    off = (off + bytes + 255) & ~(size_t)255;
    return p;
  };
  _Float16* xh    = (_Float16*)carve((size_t)SEQ_LEN * D_INNER * 2);  // 8 MB
  _Float16* winh  = (_Float16*)carve((size_t)N_DBC * D_INNER * 2);
  _Float16* wdth  = (_Float16*)carve((size_t)D_INNER * DT_RANK * 2);
  float*    dbc_f = (float*)   carve((size_t)SEQ_LEN * N_DBC * 4);
  _Float16* dbc_h = (_Float16*)carve((size_t)SEQ_LEN * N_DBC * 2);
  float*    delta = (float*)   carve((size_t)SEQ_LEN * D_INNER * 4); // 16 MB

  ssm_cvt_f16_kernel<<<512, 256, 0, stream>>>(x, W_in, W_dt, xh, winh, wdth);
  ssm_gemm_inproj_kernel<<<96, 256, 0, stream>>>(xh, winh, dbc_f, dbc_h);
  ssm_gemm_delta_kernel<<<2048, 256, 0, stream>>>(dbc_h, wdth, b_dt, delta);
  ssm_scan_kernel<<<128, 256, 0, stream>>>(x, dbc_f, delta, A_log, D, out);
}
